// f_ex_18880676233393
// MI455X (gfx1250) — compile-verified, matrix operand order checked by reading, not run on hardware
//
#include <hip/hip_runtime.h>
#include <hip/hip_bf16.h>
#include <math.h>

// ---------------------------------------------------------------------------
// 2-D DCT crop (32,3,512,512) -> (32,128) for MI455X (gfx1250, wave32, WMMA).
// Only flat DCT indices 0..1023 are gathered => only k-rows 0..15 of the
// 64x64 DCT grid are needed. Both contractions run on
// v_wmma_f32_16x16x32_bf16; x is read once (HBM-bound, ~4.3 us floor).
// This revision: double-buffered LDS X tiles (1 barrier/k-step, global loads
// overlap WMMA), D slab resident in LDS, packed bf16x2 ds_store_b32 fills.
// ---------------------------------------------------------------------------

typedef __bf16 bf16_t;
typedef bf16_t v16bf __attribute__((ext_vector_type(16)));
typedef float  v8f   __attribute__((ext_vector_type(8)));

struct alignas(16) bf16x8 { bf16_t v[8]; };

#define HW      512
#define KROWS   16      // DCT rows actually consumed by DCT_TABLE
#define LCOLS   64
#define BANDS   4
#define BANDW   128     // w columns per block
#define NBATCH  32
#define NCHAN   3
#define MSTRIDE (KROWS * LCOLS)   // 1024 floats per batch accumulator
#define XLD     40      // XsT leading dim (80 B rows -> 16B aligned)
#define DLD     520     // Ds  leading dim (1040 B rows -> 16B aligned)
#define TLD     136     // Ts  leading dim (272 B rows -> 16B aligned)

__device__ __forceinline__ unsigned bfbits(float f) {
    union { float f; unsigned u; } x; x.f = f;
    return (x.u + 0x7FFFu + ((x.u >> 16) & 1u)) >> 16;   // RNE to bf16 bits
}
__device__ __forceinline__ bf16_t f2bf(float f) {
    unsigned short h = (unsigned short)bfbits(f);
    return __builtin_bit_cast(bf16_t, h);
}
__device__ __forceinline__ unsigned pack2(float lo, float hi) {
    return (bfbits(lo) & 0xFFFFu) | (bfbits(hi) << 16);
}

// A-fragment (16x32 bf16), source row-major [M][K] with leading dim ld.
// ISA 7.12.2: lanes 0-15 row M=lane hold K {0..7,16..23}; lanes 16-31 hold
// K {8..15,24..31}. Two 16B chunk loads per lane.
__device__ __forceinline__ v16bf load_frag_a(const bf16_t* __restrict__ src,
                                             int ld, int m0, int k0) {
    const int lane = threadIdx.x & 31;
    const int m    = lane & 15;
    const int ks   = (lane >> 4) << 3;           // 0 or 8
    const bf16_t* p = src + (size_t)(m0 + m) * ld + k0 + ks;
    bf16x8 lo = *(const bf16x8*)p;
    bf16x8 hi = *(const bf16x8*)(p + 16);
    v16bf a;
#pragma unroll
    for (int i = 0; i < 8; ++i) { a[i] = lo.v[i]; a[i + 8] = hi.v[i]; }
    return a;
}

// B-fragment (32x16 bf16) where the SOURCE is laid out [N][K] (row n = output
// column, contiguous K). Mirrors the A lane split, per-lane 16B chunks.
__device__ __forceinline__ v16bf load_frag_b_nk(const bf16_t* __restrict__ src,
                                                int ld, int n0, int k0) {
    const int lane = threadIdx.x & 31;
    const int n    = lane & 15;
    const int ks   = (lane >> 4) << 3;
    const bf16_t* p = src + (size_t)(n0 + n) * ld + k0 + ks;
    bf16x8 lo = *(const bf16x8*)p;
    bf16x8 hi = *(const bf16x8*)(p + 16);
    v16bf b;
#pragma unroll
    for (int i = 0; i < 8; ++i) { b[i] = lo.v[i]; b[i + 8] = hi.v[i]; }
    return b;
}

__device__ __forceinline__ v8f wmma_bf16(v16bf a, v16bf b, v8f c) {
    return __builtin_amdgcn_wmma_f32_16x16x32_bf16(
        /*neg_a=*/false, a, /*neg_b=*/false, b,
        /*c_mod=*/(short)0, c, /*reuse_a=*/false, /*reuse_b=*/false);
}

// --------------------------- D matrix build --------------------------------
// D[k][h] = 2*cos(pi*(2h+1)*k/(2*512)), 64x512 bf16 (rows 0..15 feed stage A,
// all 64 rows feed stage B).
__global__ __launch_bounds__(256) void build_D(bf16_t* __restrict__ D) {
    int i = blockIdx.x * 256 + threadIdx.x;
    if (i < LCOLS * HW) {
        int k = i >> 9, h = i & (HW - 1);
        float v = 2.0f * cosf((float)M_PI * (2.0f * h + 1.0f) * (float)k
                              * (1.0f / (2.0f * HW)));
        D[i] = f2bf(v);
    }
}

// ------------------------------ main GEMM ----------------------------------
// One block per (image, 128-wide w band). 256 threads = 8 waves.
__global__ __launch_bounds__(256) void dct_gemm(const float* __restrict__ x,
                                                const bf16_t* __restrict__ D,
                                                float* __restrict__ m) {
    __shared__ bf16_t XsT[2][BANDW][XLD]; // X tile transposed [w][h], dbl-buf
    __shared__ bf16_t Ds[KROWS][DLD];     // D rows 0..15, all 512 cols
    __shared__ bf16_t Ts[KROWS][TLD];     // stage-A result T[k][w]

    const int band = blockIdx.x & (BANDS - 1);
    const int bc   = blockIdx.x >> 2;          // 0..95
    const int b    = bc / NCHAN;
    const int w0   = band * BANDW;
    const float* __restrict__ xim = x + (size_t)bc * HW * HW;

    const int t    = threadIdx.x;
    const int wave = t >> 5;
    const int lane = t & 31;

    // Preload D[0:16, 0:512] into LDS once: steady-state loop then touches
    // global memory only for x.
    {
        const int r = t >> 4;                  // 0..15
        const int c = (t & 15) * 32;           // 0..480
        const bf16x8* src = (const bf16x8*)(D + (size_t)r * HW + c);
        bf16x8*       dst = (bf16x8*)(&Ds[r][c]);
#pragma unroll
        for (int q = 0; q < 4; ++q) dst[q] = src[q];
    }

    // ---- Stage A: T[0:16, band] = D[0:16,:] * X[:, band] ------------------
    // Per k-step: stage a 32h x 128w f32 tile. Each thread owns an h-pair
    // (contiguous in XsT rows) x 8 w, so fills are packed ds_store_b32.
    const int hp = t >> 4;                     // 0..15: h-pair within h-tile
    const int wq = (t & 15) * 8;               // 8 w columns of the band

    float4 r0a, r0b, r1a, r1b;                 // staged next tile (registers)
    {
        const float* base = xim + (size_t)(2 * hp) * HW + w0 + wq;
        const float4* s0 = (const float4*)base;
        const float4* s1 = (const float4*)(base + HW);
        r0a = s0[0]; r0b = s0[1]; r1a = s1[0]; r1b = s1[1];
    }

    v8f acc = {};
    const int n0 = wave * 16;                  // this wave's w-tile in band

    for (int it = 0; it < 16; ++it) {
        const int buf = it & 1;
        // Commit staged registers -> LDS (bf16 pairs packed along h).
#pragma unroll
        for (int j = 0; j < 4; ++j) {
            *(unsigned*)&XsT[buf][wq + j][2 * hp]     = pack2((&r0a.x)[j], (&r1a.x)[j]);
            *(unsigned*)&XsT[buf][wq + 4 + j][2 * hp] = pack2((&r0b.x)[j], (&r1b.x)[j]);
        }
        __syncthreads();   // single barrier per k-step (double-buffered)
        // Issue next tile's global loads before consuming this one: the
        // global_load_b128 clause overlaps the WMMA + LDS reads below.
        if (it + 1 < 16) {
            const float* base = xim + (size_t)((it + 1) * 32 + 2 * hp) * HW + w0 + wq;
            const float4* s0 = (const float4*)base;
            const float4* s1 = (const float4*)(base + HW);
            r0a = s0[0]; r0b = s0[1]; r1a = s1[0]; r1b = s1[1];
            __builtin_prefetch(base + 32 * HW, 0, 1);  // global_prefetch_b8
        }
        v16bf afrag = load_frag_a(&Ds[0][0], DLD, 0, it * 32);
        v16bf bfrag = load_frag_b_nk(&XsT[buf][0][0], XLD, n0, 0);
        acc = wmma_bf16(afrag, bfrag, acc);
    }

    // Spill T tile (f32 acc -> bf16) into LDS. C/D layout: M = r + 8*(lane>=16).
    {
        const int n  = lane & 15;
        const int mh = (lane >> 4) * 8;
#pragma unroll
        for (int r = 0; r < 8; ++r)
            Ts[mh + r][n0 + n] = f2bf(acc[r]);
    }
    __syncthreads();

    // ---- Stage B: Y[0:16, 0:64] += T * D[:, band]^T -----------------------
    // 4 output l-tiles, K=128 in 4 steps; 8 waves -> each (l-tile, K-half).
    const int nt  = wave & 3;
    const int ln0 = nt * 16;
    v8f yacc = {};
#pragma unroll
    for (int s = 0; s < 2; ++s) {
        const int kstep = (wave >> 2) * 2 + s;             // 0..3
        v16bf a  = load_frag_a(&Ts[0][0], TLD, 0, kstep * 32);
        v16bf bb = load_frag_b_nk(D, HW, ln0, w0 + kstep * 32); // D is [l][w]
        yacc = wmma_bf16(a, bb, yacc);
    }
    {
        const int l  = ln0 + (lane & 15);
        const int mh = (lane >> 4) * 8;
        float* __restrict__ mb = m + (size_t)b * MSTRIDE;
#pragma unroll
        for (int r = 0; r < 8; ++r)
            atomicAdd(&mb[(mh + r) * LCOLS + l], yacc[r] * (1.0f / 3.0f));
    }
}

// ------------------------------ finalize -----------------------------------
// Regenerate DCT_TABLE (values are flat indices < 1024), gather, mean-of-8.
__global__ __launch_bounds__(128) void dct_finalize(const float* __restrict__ m,
                                                    float* __restrict__ out) {
    __shared__ int tab[1024];
    if (threadIdx.x == 0) {
        int idx = 0;
        for (int s = 0; s < 63; ++s) {
            int rlo = (s - 31 > 0) ? (s - 31) : 0;
            int rhi = (s < 31) ? s : 31;
            for (int r = rlo; r <= rhi; ++r)
                tab[idx++] = r * 32 + (s - r);
        }
    }
    __syncthreads();
    const int b = blockIdx.x, j = threadIdx.x;
    const float* __restrict__ mb = m + (size_t)b * MSTRIDE;
    float s = 0.0f;
#pragma unroll
    for (int q = 0; q < 8; ++q)
        s += mb[tab[j * 8 + q]];
    out[b * 128 + j] = s * 0.125f;
}

// ---------------------------------------------------------------------------
extern "C" void kernel_launch(void* const* d_in, const int* in_sizes, int n_in,
                              void* d_out, int out_size, void* d_ws, size_t ws_size,
                              hipStream_t stream) {
    (void)in_sizes; (void)n_in; (void)out_size; (void)ws_size;
    const float* x   = (const float*)d_in[0];
    float*       out = (float*)d_out;

    bf16_t* D = (bf16_t*)d_ws;                                  // 64*512*2 = 64 KB
    float*  m = (float*)((char*)d_ws + LCOLS * HW * sizeof(bf16_t)); // 32 KB * 4

    hipMemsetAsync(m, 0, (size_t)NBATCH * MSTRIDE * sizeof(float), stream);
    build_D<<<(LCOLS * HW + 255) / 256, 256, 0, stream>>>(D);
    dct_gemm<<<NBATCH * NCHAN * BANDS, 256, 0, stream>>>(x, D, m);
    dct_finalize<<<NBATCH, 128, 0, stream>>>(m, out);
}
